// DCENodeClassifier_10685878633295
// MI455X (gfx1250) — matrix operand order repaired and checked by reading, not compiled, not run on hardware
//
#include <hip/hip_runtime.h>
#include <hip/hip_bf16.h>

// GraphSAGE (mean) x2 + linear head for MI455X (gfx1250, wave32, WMMA).
//
// Pipeline (all on `stream`):
//   memset(deg|agg)                     -- graph-capturable memset node
//   prepack_w: Wl1,Wr1,Wl2,Wr2 (f32) -> f16 B-fragment layout (one 32B load/lane)
//   degree_kernel: deg[dst] += 1         (atomic f32, L2-resident)
//   aggregate: agg[dst] += x[src]        (1 wave/edge, 2 ch/lane, atomic f32)
//   sage_gemm<false>: h1 = relu(agg/deg @ Wl1 + bl1 + x @ Wr1)    [16 WMMA/wave]
//   memset(agg); aggregate(h1)
//   sage_gemm<true>:  out = relu(...)@Wc + bc fused via shfl_xor reduction

typedef __attribute__((ext_vector_type(16))) _Float16 v16h;
typedef __attribute__((ext_vector_type(8)))  float    v8f;

#define IN_C 64

// ---------------------------------------------------------------- degree
__global__ void degree_kernel(const int* __restrict__ dst,
                              float* __restrict__ deg, int nE) {
  int e = blockIdx.x * blockDim.x + threadIdx.x;
  if (e < nE) atomicAdd(&deg[dst[e]], 1.0f);
}

// ------------------------------------------------------------- aggregate
// One wave per edge; lane handles channels {2*lane, 2*lane+1}.
// x and agg are L2-resident (25.6 MB each << 192 MB L2).
__global__ void aggregate_kernel(const float* __restrict__ X,
                                 const int* __restrict__ src,
                                 const int* __restrict__ dst,
                                 float* __restrict__ agg, int nE) {
  long t = (long)blockIdx.x * blockDim.x + threadIdx.x;
  int e = (int)(t >> 5);
  if (e >= nE) return;
  int lane = threadIdx.x & 31;
  int s = src[e];
  int d = dst[e];
  float2 v = *(const float2*)(X + (size_t)s * IN_C + 2 * lane);
  float* a = agg + (size_t)d * IN_C + 2 * lane;
  atomicAdd(a, v.x);
  atomicAdd(a + 1, v.y);
}

// ------------------------------------------------- weight fragment prepack
// B fragment (32x16, f16): N=lane&15, g=lane>>4; halves 0..7 = K k0+g*8..+7,
// halves 8..15 = K k0+16+g*8..+7 (matches A fragment order).
// Packed as P[(mat*8 + frag)*32 + lane][16 halves], frag = (k0/32)*4 + n0/16.
__global__ void prepack_w(const float* __restrict__ W0, const float* __restrict__ W1,
                          const float* __restrict__ W2, const float* __restrict__ W3,
                          _Float16* __restrict__ P) {
  int w = threadIdx.x >> 5;          // 0..31: matrix*8 + frag
  int lane = threadIdx.x & 31;
  const float* W = (w < 8) ? W0 : (w < 16) ? W1 : (w < 24) ? W2 : W3;
  int f  = w & 7;
  int k0 = (f >> 2) * 32;
  int n0 = (f & 3) * 16;
  int g  = lane >> 4;
  int n  = lane & 15;
  const float* p = W + (size_t)(k0 + g * 8) * IN_C + n0 + n;  // column of W
  _Float16* d = P + ((size_t)w * 32 + lane) * 16;
#pragma unroll
  for (int i = 0; i < 16; ++i) d[i] = (_Float16)(p[(size_t)(i < 8 ? i : i + 8) * IN_C]);
}

// --------------------------------------------------------- A fragment load
// A (16x32 f16) per ISA 7.12.2: lane groups hold M=lane&15; halves 0..7 =
// K {k0+g*8 .. +7}, halves 8..15 = K {k0+16+g*8 .. +7}. Two 32B runs/lane.
__device__ __forceinline__ v16h load_a_frag(const float* __restrict__ S,
                                            int row, int k0, int g, float scale) {
  const float* p = S + (size_t)row * IN_C + k0 + g * 8;
  v16h a;
#pragma unroll
  for (int i = 0; i < 8; ++i) a[i] = (_Float16)(p[i] * scale);
#pragma unroll
  for (int i = 0; i < 8; ++i) a[8 + i] = (_Float16)(p[16 + i] * scale);
  return a;
}

// ------------------------------------------------------------------ GEMM
// One wave -> 16 node rows x 64 outputs. acc[t] = 16x16 C fragment for
// columns t*16..t*16+15. D = (agg*invdeg)@Wl + bl + X@Wr, relu; FINAL also
// fuses the classifier dot with W_c via a 16-lane shfl_xor butterfly.
template <bool FINAL>
__global__ __launch_bounds__(256) void sage_gemm(
    const float* __restrict__ AGG, const float* __restrict__ deg,
    const float* __restrict__ X,
    const _Float16* __restrict__ Wlp, const float* __restrict__ bl,
    const _Float16* __restrict__ Wrp,
    const float* __restrict__ Wc, const float* __restrict__ bc,
    float* __restrict__ H, float* __restrict__ out, int nN) {
  int gw   = (int)((blockIdx.x * blockDim.x + threadIdx.x) >> 5);
  int lane = threadIdx.x & 31;
  int row0 = gw * 16;
  if (row0 + 16 > nN) return;          // N is a multiple of 16 -> no remainder
  int g = lane >> 4;
  int m = lane & 15;
  int rowA = row0 + m;
  float scale = 1.0f / fmaxf(deg[rowA], 1.0f);   // mean agg, safe for deg==0

  v8f acc[4];
#pragma unroll
  for (int t = 0; t < 4; ++t) {
    float bv = bl[t * 16 + m];         // C layout: N = lane&15
#pragma unroll
    for (int j = 0; j < 8; ++j) acc[t][j] = bv;
  }

#pragma unroll
  for (int kk = 0; kk < 2; ++kk) {
    int k0 = kk * 32;
    v16h a1 = load_a_frag(AGG, rowA, k0, g, scale);
    v16h a2 = load_a_frag(X,   rowA, k0, g, 1.0f);
#pragma unroll
    for (int t = 0; t < 4; ++t) {
      v16h bL = *(const v16h*)(Wlp + (((size_t)kk * 4 + t) * 32 + lane) * 16);
      acc[t] = __builtin_amdgcn_wmma_f32_16x16x32_f16(
          false, a1, false, bL, (short)0, acc[t], false, false);
      v16h bR = *(const v16h*)(Wrp + (((size_t)kk * 4 + t) * 32 + lane) * 16);
      acc[t] = __builtin_amdgcn_wmma_f32_16x16x32_f16(
          false, a2, false, bR, (short)0, acc[t], false, false);
    }
  }

  if (!FINAL) {
    // C layout: VGPR j -> row M=j (lanes 0-15) / M=j+8 (lanes 16-31)
#pragma unroll
    for (int t = 0; t < 4; ++t)
#pragma unroll
      for (int j = 0; j < 8; ++j)
        H[(size_t)(row0 + g * 8 + j) * IN_C + t * 16 + m] =
            fmaxf(acc[t][j], 0.0f);
  } else {
    // out[row] = sum_n relu(h2[row][n]) * Wc[n] + bc
    float part[8];
#pragma unroll
    for (int j = 0; j < 8; ++j) part[j] = 0.0f;
#pragma unroll
    for (int t = 0; t < 4; ++t) {
      float wc = Wc[t * 16 + m];
#pragma unroll
      for (int j = 0; j < 8; ++j) part[j] += fmaxf(acc[t][j], 0.0f) * wc;
    }
#pragma unroll
    for (int s = 1; s < 16; s <<= 1)       // butterfly within 16-lane group
#pragma unroll
      for (int j = 0; j < 8; ++j) part[j] += __shfl_xor(part[j], s, 32);
    if (m == 0) {
      float bb = bc[0];
#pragma unroll
      for (int j = 0; j < 8; ++j) out[row0 + g * 8 + j] = part[j] + bb;
    }
  }
}

static inline size_t align_up(size_t v, size_t a) { return (v + a - 1) / a * a; }

extern "C" void kernel_launch(void* const* d_in, const int* in_sizes, int n_in,
                              void* d_out, int out_size, void* d_ws, size_t ws_size,
                              hipStream_t stream) {
  const float* x   = (const float*)d_in[0];
  const int*   ei  = (const int*)  d_in[1];
  const float* Wl1 = (const float*)d_in[2];
  const float* bl1 = (const float*)d_in[3];
  const float* Wr1 = (const float*)d_in[4];
  const float* Wl2 = (const float*)d_in[5];
  const float* bl2 = (const float*)d_in[6];
  const float* Wr2 = (const float*)d_in[7];
  const float* Wc  = (const float*)d_in[8];
  const float* bc  = (const float*)d_in[9];
  float* out = (float*)d_out;

  const int nN = in_sizes[0] / IN_C;       // 100000
  const int nE = in_sizes[1] / 2;          // 1600000
  const int* src = ei;
  const int* dst = ei + nE;

  // Workspace: deg[N] | agg[N*64] | h1[N*64] | packed f16 weights (32 KB)
  float* ws = (float*)d_ws;
  size_t aggOff = align_up((size_t)nN, 64);
  size_t h1Off  = aggOff + (size_t)nN * IN_C;
  size_t pwOff  = h1Off + (size_t)nN * IN_C;       // 32B-aligned (multiples of 64 floats)
  float* deg = ws;
  float* agg = ws + aggOff;
  float* h1  = ws + h1Off;
  _Float16* pw = (_Float16*)(ws + pwOff);          // 4 mats * 8 frags * 512 halves
  _Float16* pWl1 = pw + 0 * 4096;
  _Float16* pWr1 = pw + 1 * 4096;
  _Float16* pWl2 = pw + 2 * 4096;
  _Float16* pWr2 = pw + 3 * 4096;

  const int tpb = 256;
  const unsigned edgeBlocks = (unsigned)((nE + tpb - 1) / tpb);
  const unsigned aggBlocks  = (unsigned)(((long)nE * 32 + tpb - 1) / tpb);
  const int nTiles = nN / 16;
  const unsigned gemmBlocks = (unsigned)(((long)nTiles * 32 + tpb - 1) / tpb);

  // zero deg (+pad) and agg in one capturable memset
  hipMemsetAsync(deg, 0, sizeof(float) * (aggOff + (size_t)nN * IN_C), stream);
  prepack_w<<<1, 1024, 0, stream>>>(Wl1, Wr1, Wl2, Wr2, pw);
  degree_kernel<<<edgeBlocks, tpb, 0, stream>>>(dst, deg, nE);

  // ---- layer 1
  aggregate_kernel<<<aggBlocks, tpb, 0, stream>>>(x, src, dst, agg, nE);
  sage_gemm<false><<<gemmBlocks, tpb, 0, stream>>>(
      agg, deg, x, pWl1, bl1, pWr1, nullptr, nullptr, h1, nullptr, nN);

  // ---- layer 2 + fused classifier
  hipMemsetAsync(agg, 0, sizeof(float) * (size_t)nN * IN_C, stream);
  aggregate_kernel<<<aggBlocks, tpb, 0, stream>>>(h1, src, dst, agg, nE);
  sage_gemm<true><<<gemmBlocks, tpb, 0, stream>>>(
      agg, deg, h1, pWl2, bl2, pWr2, Wc, bc, nullptr, out, nN);
}
